// RIM_80530636800104
// MI455X (gfx1250) — matrix operand order repaired and checked by reading, not compiled, not run on hardware
//
#include <hip/hip_runtime.h>
#include <hip/hip_bf16.h>
#include <math.h>

#define B_ 64
#define T_ 196
#define U_ 6
#define H_ 512
#define ACTIVE_ 4
#define KIN_ 64
#define CH_ 4
#define CK_ 32
#define UH_ (U_ * H_)        /* 3072 */
#define CD_ (CH_ * H_)       /* 2048 */
#define COR_ (CH_ * U_ * H_) /* 12288 */
#define UK_ (U_ * CH_ * CK_) /* 768 */

typedef __attribute__((ext_vector_type(16))) _Float16 v16h;
typedef __attribute__((ext_vector_type(8)))  float    v8f;
typedef __attribute__((ext_vector_type(4)))  int      v4i;

#if defined(__HIP_DEVICE_COMPILE__) && __has_builtin(__builtin_amdgcn_global_load_async_to_lds_b128)
#define HAVE_ASYNC_LDS 1
#else
#define HAVE_ASYNC_LDS 0
#endif

#define AS1 __attribute__((address_space(1)))
#define AS3 __attribute__((address_space(3)))

__device__ __forceinline__ void wait_async_copies() {
#if HAVE_ASYNC_LDS
#if __has_builtin(__builtin_amdgcn_s_wait_asynccnt)
  __builtin_amdgcn_s_wait_asynccnt(0);
#else
  asm volatile("s_wait_asynccnt 0x0" ::: "memory");
#endif
#endif
}

// Stage a 16-row x K-col f16 panel (row stride lda) into LDS, contiguous rows of K.
// Uses gfx1250 async global->LDS copies when available; falls back to vector copies.
__device__ __forceinline__ void stage_a16(_Float16* lds, const _Float16* A, int lda, int K) {
  int kc = K >> 3;              // 16B chunks per row
  int chunks = kc << 4;         // 16 rows
  for (int i = threadIdx.x; i < chunks; i += blockDim.x) {
    int r = i / kc, c = (i - r * kc) << 3;
#if HAVE_ASYNC_LDS
    __builtin_amdgcn_global_load_async_to_lds_b128(
        (AS1 v4i*)(AS1 void*)(A + (size_t)r * lda + c),
        (AS3 v4i*)(AS3 void*)(lds + r * K + c), 0, 0);
#else
    *reinterpret_cast<float4*>(lds + r * K + c) =
        *reinterpret_cast<const float4*>(A + (size_t)r * lda + c);
#endif
  }
  wait_async_copies();
  __syncthreads();
}

// ---------------- WMMA fragment loaders (CDNA5 wave32 layouts) ----------------
// A 16x32 f16: lane L holds row M=L&15; VGPR j holds K pair:
//   K = (j>=4 ? 16:0) + (j&3)*2 + (L>=16 ? 8:0)
__device__ __forceinline__ v16h load_a_frag(const _Float16* A, int lda) {
  int lane = threadIdx.x & 31;
  int r = lane & 15, hi = lane >> 4;
  const _Float16* row = A + r * lda;
  unsigned tmp[8];
#pragma unroll
  for (int j = 0; j < 8; ++j) {
    int kb = ((j & 4) << 2) + ((j & 3) << 1) + (hi << 3);
    tmp[j] = *reinterpret_cast<const unsigned*>(row + kb);
  }
  v16h out;
  __builtin_memcpy(&out, tmp, sizeof(out));
  return out;
}

// B 32x16 f16 (row-major K x N): lane L holds column N=L&15; element s holds
//   K = (s>=8 ? 16:0) + (s&7) + (L>=16 ? 8:0)
__device__ __forceinline__ v16h load_b_frag(const _Float16* Bm, int ldb) {
  int lane = threadIdx.x & 31;
  int n = lane & 15, hi = lane >> 4;
  v16h out;
#pragma unroll
  for (int s = 0; s < 16; ++s) {
    int k = ((s & 8) << 1) + (hi << 3) + (s & 7);
    out[s] = Bm[k * ldb + n];
  }
  return out;
}

#define WMMA_F16(a, b, c) \
  __builtin_amdgcn_wmma_f32_16x16x32_f16(false, (a), false, (b), (short)0, (c), false, false)

// One wave: 16(M) x 64(N) tile. A read from LDS (shared per block), B streamed from
// global with register double-buffering so WMMAs overlap in-flight loads.
__device__ __forceinline__ void wave_gemm_ldsA_16x64(const _Float16* Alds,
                                                     const _Float16* Bm, int ldb,
                                                     int K, v8f acc[4]) {
  v16h b0 = load_b_frag(Bm, ldb);
  v16h b1 = load_b_frag(Bm + 16, ldb);
  v16h b2 = load_b_frag(Bm + 32, ldb);
  v16h b3 = load_b_frag(Bm + 48, ldb);
  int kk = 0;
  for (; kk + 32 < K; kk += 32) {
    const _Float16* Bn = Bm + (kk + 32) * ldb;
    if (kk + 64 < K) __builtin_prefetch(Bn + 32 * ldb, 0, 1);  // L2 prefetch 2 ahead
    v16h nb0 = load_b_frag(Bn, ldb);
    v16h nb1 = load_b_frag(Bn + 16, ldb);
    v16h nb2 = load_b_frag(Bn + 32, ldb);
    v16h nb3 = load_b_frag(Bn + 48, ldb);
    v16h a = load_a_frag(Alds + kk, K);
    acc[0] = WMMA_F16(a, b0, acc[0]);
    acc[1] = WMMA_F16(a, b1, acc[1]);
    acc[2] = WMMA_F16(a, b2, acc[2]);
    acc[3] = WMMA_F16(a, b3, acc[3]);
    b0 = nb0; b1 = nb1; b2 = nb2; b3 = nb3;
  }
  v16h a = load_a_frag(Alds + kk, K);
  acc[0] = WMMA_F16(a, b0, acc[0]);
  acc[1] = WMMA_F16(a, b1, acc[1]);
  acc[2] = WMMA_F16(a, b2, acc[2]);
  acc[3] = WMMA_F16(a, b3, acc[3]);
}

// ---------------- prep kernels ----------------
__global__ void k_f32_to_f16(const float* __restrict__ src,
                             _Float16* __restrict__ dst, int n) {
  int i = blockIdx.x * blockDim.x + threadIdx.x;
  if (i < n) dst[i] = (_Float16)src[i];
}

// wqa[u,h] = sum_d iqw[u,h,d]*ikw[d]; wqb with input_key_b
__global__ void k_prep_gate_w(const float* __restrict__ iqw, const float* __restrict__ ikw,
                              const float* __restrict__ ikb,
                              float* __restrict__ wqa, float* __restrict__ wqb) {
  int idx = blockIdx.x * blockDim.x + threadIdx.x;  // u*H + h
  if (idx >= UH_) return;
  const float* row = iqw + (size_t)idx * KIN_;
  float a = 0.f, b = 0.f;
  for (int d = 0; d < KIN_; ++d) { a += row[d] * ikw[d]; b += row[d] * ikb[d]; }
  wqa[idx] = a; wqb[idx] = b;
}

// winv[u,h] = sum_d ivw[d]*win[u,d,h]; winb with input_value_b
__global__ void k_prep_win(const float* __restrict__ win, const float* __restrict__ ivw,
                           const float* __restrict__ ivb,
                           float* __restrict__ winv, float* __restrict__ winb) {
  int idx = blockIdx.x * blockDim.x + threadIdx.x;  // u*H + h
  if (idx >= UH_) return;
  int u = idx / H_, h = idx % H_;
  const float* base = win + (size_t)u * KIN_ * H_ + h;
  float a = 0.f, b = 0.f;
  for (int d = 0; d < KIN_; ++d) { float w = base[d * H_]; a += ivw[d] * w; b += ivb[d] * w; }
  winv[idx] = a; winb[idx] = b;
}

__global__ void k_init_hs(float* __restrict__ hs, _Float16* __restrict__ hs_h) {
  int i = blockIdx.x * blockDim.x + threadIdx.x;
  if (i < B_ * UH_) { hs[i] = 0.001f; hs_h[i] = (_Float16)0.001f; }
}

// ---------------- per-step kernels ----------------
// Input attention collapsed to rank-1 form + exact top-4 mask.
__global__ void k_gate(const float* __restrict__ hs, const float* __restrict__ wqa,
                       const float* __restrict__ wqb, const float* __restrict__ x, int t,
                       float* __restrict__ mask, float* __restrict__ coef) {
  int b = blockIdx.x;
  int tid = threadIdx.x, warp = tid >> 5, lane = tid & 31;
  __shared__ float sqa[U_], sqb[U_];
  if (warp < U_) {
    const float* hrow = hs + ((size_t)b * U_ + warp) * H_;
    const float* wa = wqa + warp * H_;
    const float* wb = wqb + warp * H_;
    float pa = 0.f, pb = 0.f;
    for (int h = lane; h < H_; h += 32) { float hv = hrow[h]; pa += hv * wa[h]; pb += hv * wb[h]; }
    for (int off = 16; off; off >>= 1) { pa += __shfl_down(pa, off); pb += __shfl_down(pb, off); }
    if (lane == 0) { sqa[warp] = pa; sqb[warp] = pb; }
  }
  __syncthreads();
  if (tid == 0) {
    float xt = x[b * T_ + t];
    float s0v[U_], nullv[U_];
    for (int u = 0; u < U_; ++u) {
      float l0 = (xt * sqa[u] + sqb[u]) * 0.125f;  // 1/sqrt(KIN)
      float l1 = sqb[u] * 0.125f;
      float m = fmaxf(l0, l1);
      float e0 = __expf(l0 - m), e1 = __expf(l1 - m);
      s0v[u] = e0 / (e0 + e1);
      nullv[u] = e1 / (e0 + e1);
    }
    float mk[U_];
    for (int u = 0; u < U_; ++u) mk[u] = 0.f;
    for (int k = 0; k < ACTIVE_; ++k) {  // 4 smallest null scores, ties -> lowest index
      int best = 0; float bv = 3.402823466e38f;
      for (int u = 0; u < U_; ++u)
        if (mk[u] == 0.f && nullv[u] < bv) { bv = nullv[u]; best = u; }
      mk[best] = 1.f;
    }
    for (int u = 0; u < U_; ++u) {
      mask[b * U_ + u] = mk[u];
      coef[b * U_ + u] = mk[u] * s0v[u] * xt;
    }
  }
}

// pre = hs@wrec + bres + coef*winv + mask*winb ; h_bg = 0.5*hs + 0.5*tanh(pre)
// grid(4 mtiles, U); block = 8 waves = 8 N-strips sharing one LDS A panel.
__global__ void k_pre_gemm(const _Float16* __restrict__ hs_h, const _Float16* __restrict__ wrec_h,
                           const float* __restrict__ hs, const float* __restrict__ bres,
                           const float* __restrict__ winv, const float* __restrict__ winb,
                           const float* __restrict__ mask, const float* __restrict__ coef,
                           float* __restrict__ hbg, _Float16* __restrict__ hbg_h) {
  __shared__ _Float16 Alds[16 * H_];  // 16 KB
  int u = blockIdx.y;
  int m0 = blockIdx.x * 16;
  stage_a16(Alds, hs_h + (size_t)m0 * UH_ + u * H_, UH_, H_);
  int nstrip = threadIdx.x >> 5;
  int n0 = nstrip * 64;
  const _Float16* Bm = wrec_h + (size_t)u * H_ * H_ + n0;
  v8f acc[4] = {};
  wave_gemm_ldsA_16x64(Alds, Bm, H_, H_, acc);
  int lane = threadIdx.x & 31;
  int ncol = lane & 15, mhi = (lane >> 4) * 8;
#pragma unroll
  for (int t = 0; t < 4; ++t) {
    int n = n0 + 16 * t + ncol;
    float bv = bres[u * H_ + n], wv = winv[u * H_ + n], wb = winb[u * H_ + n];
#pragma unroll
    for (int r = 0; r < 8; ++r) {
      int b = m0 + mhi + r;
      size_t oi = ((size_t)b * U_ + u) * H_ + n;
      float val = acc[t][r] + bv + coef[b * U_ + u] * wv + mask[b * U_ + u] * wb;
      float hr = 0.5f * hs[oi] + 0.5f * tanhf(val);
      hbg[oi] = hr;
      hbg_h[oi] = (_Float16)hr;
    }
  }
}

// ck/cq = h_bg @ comm_{key,query}_w[u] : 64x512x128 each.
// grid(4 mtiles, U); block = 4 waves: {key,query} x {n0,n64}, sharing one LDS A panel.
__global__ void k_ckq_gemm(const _Float16* __restrict__ hbg_h, const _Float16* __restrict__ ckw_h,
                           const _Float16* __restrict__ cqw_h, float* __restrict__ ckmat,
                           float* __restrict__ cqmat) {
  __shared__ _Float16 Alds[16 * H_];  // 16 KB
  int u = blockIdx.y;
  int m0 = blockIdx.x * 16;
  stage_a16(Alds, hbg_h + (size_t)m0 * UH_ + u * H_, UH_, H_);
  int wave = threadIdx.x >> 5;
  int sel = wave >> 1, nstrip = wave & 1;
  const _Float16* W = (sel ? cqw_h : ckw_h) + (size_t)u * H_ * 128;
  float* out = sel ? cqmat : ckmat;
  int n0 = nstrip * 64;
  v8f acc[4] = {};
  wave_gemm_ldsA_16x64(Alds, W + n0, 128, H_, acc);
  int lane = threadIdx.x & 31;
  int ncol = lane & 15, mhi = (lane >> 4) * 8;
#pragma unroll
  for (int t = 0; t < 4; ++t) {
    int n = n0 + 16 * t + ncol;
#pragma unroll
    for (int r = 0; r < 8; ++r) {
      int b = m0 + mhi + r;
      out[(size_t)b * UK_ + u * 128 + n] = acc[t][r];
    }
  }
}

// comm scores softmax (masked rows -> uniform) + value mix; writes co_r f16 flat (B,12288)
__global__ void k_comm_mix(const float* __restrict__ ckmat, const float* __restrict__ cqmat,
                           const float* __restrict__ mask, const float* __restrict__ hbg,
                           _Float16* __restrict__ co_r) {
  int chu = blockIdx.x, b = blockIdx.y;
  int ch = chu / U_, u = chu % U_;
  __shared__ float cs[U_];
  int tid = threadIdx.x;
  if (tid < U_) {
    const float* qrow = cqmat + (size_t)b * UK_ + ch * 192 + u * 32;   // contiguous 32
    const float* krow = ckmat + (size_t)b * UK_ + ch * 192 + tid * 32;
    float d = 0.f;
    for (int k = 0; k < CK_; ++k) d += qrow[k] * krow[k];
    cs[tid] = d * 0.17677669529663689f * mask[b * U_ + u];  // 1/sqrt(32), cm mask
  }
  __syncthreads();
  if (tid == 0) {
    float mx = cs[0];
    for (int v = 1; v < U_; ++v) mx = fmaxf(mx, cs[v]);
    float e[U_], s = 0.f;
    for (int v = 0; v < U_; ++v) { e[v] = __expf(cs[v] - mx); s += e[v]; }
    for (int v = 0; v < U_; ++v) cs[v] = e[v] / s;
  }
  __syncthreads();
  for (int hh = tid; hh < H_; hh += blockDim.x) {
    float a = 0.f;
    for (int v = 0; v < U_; ++v) a += cs[v] * hbg[((size_t)b * U_ + v) * H_ + hh];
    co_r[(size_t)b * COR_ + ch * UH_ + u * H_ + hh] = (_Float16)a;
  }
}

// co = co_r @ comm_output_w[u] (64x2048x512) ; fused state update.
// grid(4 mtiles, U); block = 8 waves = 8 N-strips sharing one 64 KB LDS A panel.
__global__ void k_co_gemm(const _Float16* __restrict__ co_r, const _Float16* __restrict__ cow_h,
                          const float* __restrict__ hbg, const float* __restrict__ mask,
                          float* __restrict__ hs, _Float16* __restrict__ hs_h) {
  __shared__ _Float16 Alds[16 * CD_];  // 64 KB
  int u = blockIdx.y;
  int m0 = blockIdx.x * 16;
  stage_a16(Alds, co_r + (size_t)m0 * COR_ + u * CD_, COR_, CD_);
  int nstrip = threadIdx.x >> 5;
  int n0 = nstrip * 64;
  const _Float16* Bm = cow_h + (size_t)u * CD_ * H_ + n0;
  v8f acc[4] = {};
  wave_gemm_ldsA_16x64(Alds, Bm, H_, CD_, acc);
  int lane = threadIdx.x & 31;
  int ncol = lane & 15, mhi = (lane >> 4) * 8;
#pragma unroll
  for (int t = 0; t < 4; ++t) {
    int n = n0 + 16 * t + ncol;
#pragma unroll
    for (int r = 0; r < 8; ++r) {
      int b = m0 + mhi + r;
      size_t oi = ((size_t)b * U_ + u) * H_ + n;
      float hb = hbg[oi];
      float m = mask[b * U_ + u];
      float hu = 0.5f * acc[t][r] + 0.5f * hb;
      float hn = m * hu + (1.f - m) * hs[oi];
      hs[oi] = hn;
      hs_h[oi] = (_Float16)hn;
    }
  }
}

// readout: relu(hs_flat @ w1 + b1) @ w2 + b2
__global__ void k_readout(const float* __restrict__ hs, const float* __restrict__ w1,
                          const float* __restrict__ b1, const float* __restrict__ w2,
                          const float* __restrict__ b2, float* __restrict__ out) {
  int b = blockIdx.x;
  __shared__ float h1[200];
  const float* row = hs + (size_t)b * UH_;
  for (int j = threadIdx.x; j < 200; j += blockDim.x) {
    float a = b1[j];
    for (int i = 0; i < UH_; ++i) a += row[i] * w1[i * 200 + j];
    h1[j] = fmaxf(a, 0.f);
  }
  __syncthreads();
  if (threadIdx.x < 10) {
    float a = b2[threadIdx.x];
    for (int j = 0; j < 200; ++j) a += h1[j] * w2[j * 10 + threadIdx.x];
    out[b * 10 + threadIdx.x] = a;
  }
}

// ---------------- host ----------------
extern "C" void kernel_launch(void* const* d_in, const int* in_sizes, int n_in,
                              void* d_out, int out_size, void* d_ws, size_t ws_size,
                              hipStream_t stream) {
  (void)in_sizes; (void)n_in; (void)out_size; (void)ws_size;
  const float* x    = (const float*)d_in[0];
  const float* ikw  = (const float*)d_in[1];
  const float* ikb  = (const float*)d_in[2];
  const float* ivw  = (const float*)d_in[3];
  const float* ivb  = (const float*)d_in[4];
  const float* iqw  = (const float*)d_in[5];
  const float* ckw  = (const float*)d_in[6];
  const float* cqw  = (const float*)d_in[7];
  const float* cow  = (const float*)d_in[8];
  const float* win  = (const float*)d_in[9];
  const float* wrec = (const float*)d_in[10];
  const float* bres = (const float*)d_in[11];
  const float* w1   = (const float*)d_in[12];
  const float* b1   = (const float*)d_in[13];
  const float* w2   = (const float*)d_in[14];
  const float* b2   = (const float*)d_in[15];
  float* out = (float*)d_out;

  char* p = (char*)d_ws;
  auto carve = [&](size_t bytes) { char* r = p; p += (bytes + 255) & ~(size_t)255; return r; };
  _Float16* wrec_h = (_Float16*)carve(sizeof(_Float16) * (size_t)U_ * H_ * H_);
  _Float16* ckw_h  = (_Float16*)carve(sizeof(_Float16) * (size_t)U_ * H_ * 128);
  _Float16* cqw_h  = (_Float16*)carve(sizeof(_Float16) * (size_t)U_ * H_ * 128);
  _Float16* cow_h  = (_Float16*)carve(sizeof(_Float16) * (size_t)U_ * CD_ * H_);
  float* wqa  = (float*)carve(sizeof(float) * UH_);
  float* wqb  = (float*)carve(sizeof(float) * UH_);
  float* winv = (float*)carve(sizeof(float) * UH_);
  float* winb = (float*)carve(sizeof(float) * UH_);
  float*    hs    = (float*)carve(sizeof(float) * (size_t)B_ * UH_);
  _Float16* hs_h  = (_Float16*)carve(sizeof(_Float16) * (size_t)B_ * UH_);
  float*    hbg   = (float*)carve(sizeof(float) * (size_t)B_ * UH_);
  _Float16* hbg_h = (_Float16*)carve(sizeof(_Float16) * (size_t)B_ * UH_);
  float* maskb = (float*)carve(sizeof(float) * B_ * U_);
  float* coefb = (float*)carve(sizeof(float) * B_ * U_);
  float* ckmat = (float*)carve(sizeof(float) * (size_t)B_ * UK_);
  float* cqmat = (float*)carve(sizeof(float) * (size_t)B_ * UK_);
  _Float16* co_r = (_Float16*)carve(sizeof(_Float16) * (size_t)B_ * COR_);

  // weight prep (every call: deterministic, inputs never mutated)
  {
    int n;
    n = U_ * H_ * H_;
    k_f32_to_f16<<<(n + 255) / 256, 256, 0, stream>>>(wrec, wrec_h, n);
    n = U_ * H_ * 128;
    k_f32_to_f16<<<(n + 255) / 256, 256, 0, stream>>>(ckw, ckw_h, n);
    k_f32_to_f16<<<(n + 255) / 256, 256, 0, stream>>>(cqw, cqw_h, n);
    n = U_ * CD_ * H_;
    k_f32_to_f16<<<(n + 255) / 256, 256, 0, stream>>>(cow, cow_h, n);
    k_prep_gate_w<<<(UH_ + 255) / 256, 256, 0, stream>>>(iqw, ikw, ikb, wqa, wqb);
    k_prep_win<<<(UH_ + 255) / 256, 256, 0, stream>>>(win, ivw, ivb, winv, winb);
    n = B_ * UH_;
    k_init_hs<<<(n + 255) / 256, 256, 0, stream>>>(hs, hs_h);
  }

  for (int t = 0; t < T_; ++t) {
    k_gate<<<B_, 256, 0, stream>>>(hs, wqa, wqb, x, t, maskb, coefb);
    k_pre_gemm<<<dim3(4, U_), 256, 0, stream>>>(hs_h, wrec_h, hs, bres, winv, winb,
                                                maskb, coefb, hbg, hbg_h);
    k_ckq_gemm<<<dim3(4, U_), 128, 0, stream>>>(hbg_h, ckw_h, cqw_h, ckmat, cqmat);
    k_comm_mix<<<dim3(CH_ * U_, B_), 256, 0, stream>>>(ckmat, cqmat, maskb, hbg, co_r);
    k_co_gemm<<<dim3(4, U_), 256, 0, stream>>>(co_r, cow_h, hbg, maskb, hs, hs_h);
  }

  k_readout<<<B_, 256, 0, stream>>>(hs, w1, b1, w2, b2, out);
}
